// ShiftAttention_4088808866277
// MI455X (gfx1250) — matrix operand order repaired and checked by reading, not compiled, not run on hardware
//
#include <hip/hip_runtime.h>

// ---- CDNA5 WMMA types ----
typedef __attribute__((ext_vector_type(16))) __bf16 v16bf;
typedef __attribute__((ext_vector_type(8)))  __bf16 v8bf;
typedef __attribute__((ext_vector_type(8)))  float  v8f;

#define S_LEN 2048
#define D_DIM 64
#define BM    64      // query rows per workgroup (4 waves x 16)
#define BN    64      // key cols per tile
#define NWAVE 4
#define SHIFT 2
#define NEG_BIG (-1e30f)
#define NTILE (S_LEN / BN)

// low 32 bits of a generic LDS pointer == wave-relative LDS byte offset
#define LDS_OFF(p) ((unsigned)(unsigned long long)(p))

// async global->LDS, 16B per lane, tracked by ASYNCcnt
__device__ __forceinline__ void async_copy_b128(unsigned lds_off, unsigned long long gaddr) {
  asm volatile("global_load_async_to_lds_b128 %0, %1, off"
               :: "v"(lds_off), "v"(gaddr) : "memory");
}
__device__ __forceinline__ void wait_async0() {
  asm volatile("s_wait_asynccnt 0" ::: "memory");
}

// 16 contiguous bf16 -> B operand half (caller applies lane-half K offset)
__device__ __forceinline__ v16bf ld_cont16(const __bf16* p) {
  v8bf a = *(const v8bf*)p;
  v8bf b = *(const v8bf*)(p + 8);
  return __builtin_shufflevector(a, b, 0,1,2,3,4,5,6,7,8,9,10,11,12,13,14,15);
}
// A-operand pattern: elements 0..7 = p[0..7], elements 8..15 = p[16..23]
__device__ __forceinline__ v16bf ld_split16(const __bf16* p) {
  v8bf a = *(const v8bf*)p;
  v8bf b = *(const v8bf*)(p + 16);
  return __builtin_shufflevector(a, b, 0,1,2,3,4,5,6,7,8,9,10,11,12,13,14,15);
}

__global__ __launch_bounds__(128)
void ShiftAttention_flash_wmma(const float* __restrict__ Q,
                               const float* __restrict__ K,
                               const float* __restrict__ V,
                               const int*   __restrict__ M,
                               float* __restrict__ O)
{
  // double-buffered raw f32 staging filled by the async engine
  __shared__ __attribute__((aligned(16))) float  sKf[2][BN * D_DIM];   // 2 x 16KB
  __shared__ __attribute__((aligned(16))) float  sVf[2][BN * D_DIM];   // 2 x 16KB
  // bf16 operand tiles
  __shared__ __attribute__((aligned(32))) __bf16 sK [BN * D_DIM];      // [key n][d]
  __shared__ __attribute__((aligned(32))) __bf16 sVT[D_DIM * BN];      // [d][key k] transposed
  __shared__ __attribute__((aligned(32))) __bf16 sP [NWAVE][16 * BN];  // per-wave P scratch

  const int tid  = threadIdx.x;
  const int wave = tid >> 5;
  const int lane = tid & 31;
  const int l16  = lane & 15;
  const bool hi  = lane >= 16;

  const int  bh   = blockIdx.y;
  const int  q0   = blockIdx.x * BM;
  const long base = (long)bh * S_LEN * D_DIM;

  const unsigned long long khead = (unsigned long long)(K + base);  // byte addr of head slice
  const unsigned long long vhead = (unsigned long long)(V + base);
  const unsigned kf0 = LDS_OFF(&sKf[0][0]);
  const unsigned kf1 = LDS_OFF(&sKf[1][0]);
  const unsigned vf0 = LDS_OFF(&sVf[0][0]);
  const unsigned vf1 = LDS_OFF(&sVf[1][0]);
  // per-lane base byte offset within a 16KB tile
  const unsigned lane_o = (unsigned)(wave * 4096 + lane * 16);

  // ---- issue async copies of tile kb into staging buffer buf (8+8 b128 per wave) ----
  auto issue_tile = [&](int kb, int buf) {
    const unsigned klds = (buf ? kf1 : kf0);
    const unsigned vlds = (buf ? vf1 : vf0);
    const unsigned long long kbyte = (unsigned long long)kb * (D_DIM * 4);          // K tile start
    const unsigned long long vbyte = (unsigned long long)(kb + SHIFT) * (D_DIM * 4);// V rows +SHIFT
    const unsigned long long vmax  = (unsigned long long)S_LEN * (D_DIM * 4) - 16;  // clamp in-head
    #pragma unroll
    for (int i = 0; i < 8; ++i) {
      const unsigned o = lane_o + (unsigned)(i * 512);
      async_copy_b128(klds + o, khead + kbyte + o);
      unsigned long long vo = vbyte + o;
      if (vo > vmax) vo = vmax;  // tail rows are -inf masked; clamp keeps address in range
      async_copy_b128(vlds + o, vhead + vo);
    }
  };

  // ---------- Q A-operands (persistent): rows m = q0 + wave*16 + l16 ----------
  v16bf qa[2];
  {
    const float* qrow = Q + base + (long)(q0 + wave * 16 + l16) * D_DIM;
    #pragma unroll
    for (int c = 0; c < 2; ++c) {
      const int db = c * 32 + (hi ? 8 : 0);
      #pragma unroll
      for (int e = 0; e < 8; ++e) qa[c][e]     = (__bf16)qrow[db + e];
      #pragma unroll
      for (int e = 0; e < 8; ++e) qa[c][8 + e] = (__bf16)qrow[db + 16 + e];
    }
  }

  // ---------- online-softmax state ----------
  float mrow[8], lrow[8];
  v8f acc[4];
  #pragma unroll
  for (int r = 0; r < 8; ++r) { mrow[r] = NEG_BIG; lrow[r] = 0.0f; }
  #pragma unroll
  for (int t = 0; t < 4; ++t) {
    #pragma unroll
    for (int e = 0; e < 8; ++e) acc[t][e] = 0.0f;
  }

  const int qbase_row = q0 + wave * 16;

  // prologue: start streaming tile 0
  issue_tile(0, 0);

  // ================= main K/V tile loop =================
  for (int t = 0; t < NTILE; ++t) {
    const int kb  = t * BN;
    const int buf = t & 1;

    wait_async0();       // this wave's async copies for tile t done
    __syncthreads();     // everyone's done; also fences last iter's ds reads of sK/sVT

    if (t + 1 < NTILE) issue_tile(kb + BN, buf ^ 1);   // overlap t+1 stream with t compute

    // ---- convert pass: LDS f32 -> bf16 operand tiles ----
    {
      const int r = tid >> 1, h = tid & 1;
      const float* ksrc = &sKf[buf][r * D_DIM + h * 32];
      __bf16* kdst = &sK[r * D_DIM + h * 32];
      #pragma unroll
      for (int e = 0; e < 32; ++e) kdst[e] = (__bf16)ksrc[e];
      const float* vsrc = &sVf[buf][r * D_DIM + h * 32];
      #pragma unroll
      for (int e = 0; e < 32; ++e) sVT[(h * 32 + e) * BN + r] = (__bf16)vsrc[e];
    }
    __syncthreads();

    // ---- S = Q K^T  (4 col-tiles x 2 head-dim chunks of K=32) ----
    v8f st[4];
    #pragma unroll
    for (int nc = 0; nc < 4; ++nc) {
      v8f c;
      #pragma unroll
      for (int e = 0; e < 8; ++e) c[e] = 0.0f;
      #pragma unroll
      for (int dc = 0; dc < 2; ++dc) {
        const __bf16* bp = &sK[(nc * 16 + l16) * D_DIM + dc * 32 + (hi ? 8 : 0)];
        v16bf b = ld_split16(bp);
        c = __builtin_amdgcn_wmma_f32_16x16x32_bf16(false, qa[dc], false, b,
                                                    (short)0, c, false, false);
      }
      st[nc] = c;
    }

    // ---- scale + mask + roll-validity (j <= S-1-SHIFT) ----
    #pragma unroll
    for (int nc = 0; nc < 4; ++nc) {
      const int j = kb + nc * 16 + l16;
      const bool jvalid = (j <= S_LEN - 1 - SHIFT);
      #pragma unroll
      for (int r = 0; r < 8; ++r) {
        const int mq = qbase_row + r + (hi ? 8 : 0);
        const int mv = M[(long)mq * S_LEN + j];
        const float s = st[nc][r] * 0.125f;                  // 1/sqrt(64)
        st[nc][r] = (jvalid && (mv != 0)) ? s : -__builtin_inff();
      }
    }

    // ---- online softmax update ----
    #pragma unroll
    for (int r = 0; r < 8; ++r) {
      float v = fmaxf(fmaxf(st[0][r], st[1][r]), fmaxf(st[2][r], st[3][r]));
      #pragma unroll
      for (int off = 8; off >= 1; off >>= 1) v = fmaxf(v, __shfl_xor(v, off, 32));
      const float mn    = fmaxf(mrow[r], v);
      const float alpha = __expf(mrow[r] - mn);
      mrow[r] = mn;
      float s0 = 0.0f;
      #pragma unroll
      for (int nc = 0; nc < 4; ++nc) {
        const float p = __expf(st[nc][r] - mn);
        st[nc][r] = p;
        s0 += p;
      }
      #pragma unroll
      for (int off = 8; off >= 1; off >>= 1) s0 += __shfl_xor(s0, off, 32);
      lrow[r] = lrow[r] * alpha + s0;
      #pragma unroll
      for (int dc = 0; dc < 4; ++dc) acc[dc][r] *= alpha;
    }

    // ---- P (D-layout f32) -> wave-private LDS bf16, row-major 16 x BN ----
    __bf16* pw = &sP[wave][0];
    #pragma unroll
    for (int nc = 0; nc < 4; ++nc)
      #pragma unroll
      for (int r = 0; r < 8; ++r)
        pw[(r + (hi ? 8 : 0)) * BN + nc * 16 + l16] = (__bf16)st[nc][r];

    asm volatile("s_wait_dscnt 0" ::: "memory");   // wave-local LDS RAW fence

    // ---- acc += P @ V'  (4 d-tiles x 2 key chunks of K=32) ----
    v16bf pa[2];
    #pragma unroll
    for (int kc = 0; kc < 2; ++kc)
      pa[kc] = ld_split16(pw + l16 * BN + kc * 32 + (hi ? 8 : 0));
    #pragma unroll
    for (int dc = 0; dc < 4; ++dc) {
      v8f c = acc[dc];
      #pragma unroll
      for (int kc = 0; kc < 2; ++kc) {
        const __bf16* vb = &sVT[(dc * 16 + l16) * BN + kc * 32 + (hi ? 8 : 0)];
        v16bf b = ld_split16(vb);
        c = __builtin_amdgcn_wmma_f32_16x16x32_bf16(false, pa[kc], false, b,
                                                    (short)0, c, false, false);
      }
      acc[dc] = c;
    }
  }

  // ---------- epilogue: out = acc / l ----------
  #pragma unroll
  for (int r = 0; r < 8; ++r) {
    const float inv = 1.0f / lrow[r];
    const int   mq  = qbase_row + r + (hi ? 8 : 0);
    float* orow = O + base + (long)mq * D_DIM;
    #pragma unroll
    for (int dc = 0; dc < 4; ++dc)
      orow[dc * 16 + l16] = acc[dc][r] * inv;
  }
}

extern "C" void kernel_launch(void* const* d_in, const int* in_sizes, int n_in,
                              void* d_out, int out_size, void* d_ws, size_t ws_size,
                              hipStream_t stream) {
  (void)in_sizes; (void)n_in; (void)out_size; (void)d_ws; (void)ws_size;
  const float* Q = (const float*)d_in[0];
  const float* K = (const float*)d_in[1];
  const float* V = (const float*)d_in[2];
  const int*   M = (const int*)  d_in[3];
  float*       O = (float*)d_out;

  dim3 grid(S_LEN / BM, 4 * 16);   // 32 query tiles x 64 (batch*head)
  dim3 block(32 * NWAVE);          // 4 waves (wave32)
  ShiftAttention_flash_wmma<<<grid, block, 0, stream>>>(Q, K, V, M, O);
}
